// MixtralAttention_21182778704250
// MI455X (gfx1250) — matrix operand order repaired and checked by reading, not compiled, not run on hardware
//
#include <hip/hip_runtime.h>
#include <hip/hip_bf16.h>
#include <math.h>

// ---- problem constants (match reference) ----
#define HID   4096
#define SEQ   2048
#define NH    32
#define NKV   8
#define HDIM  128
#define NHHD  (NH * HDIM)    // 4096
#define NKVHD (NKV * HDIM)   // 1024

typedef __attribute__((ext_vector_type(16))) __bf16 v16bf;
typedef __attribute__((ext_vector_type(8)))  __bf16 v8bf;
typedef __attribute__((ext_vector_type(8)))  float  v8f;

union V16U { v16bf v; v8bf h[2]; };

__device__ __forceinline__ v8f zero8() {
  v8f z;
#pragma unroll
  for (int r = 0; r < 8; ++r) z[r] = 0.0f;
  return z;
}

__device__ __forceinline__ v8f wmma_bf16(v16bf a, v16bf b, v8f c) {
  // D = A(16x32) * B(32x16) + C, f32 accumulate
  return __builtin_amdgcn_wmma_f32_16x16x32_bf16(
      /*neg_a=*/false, a, /*neg_b=*/false, b,
      /*c_mod=*/(short)0, c, /*reuse_a=*/false, /*reuse_b=*/false);
}

// A fragment (16x32, row-major, leading dim lda in elements), ptr at (row0,k0).
// Layout: lanes 0-15 -> K 0..15; lanes 16-31 -> K 16..31 (matches load_b_frag).
__device__ __forceinline__ v16bf load_a_frag(const __bf16* p, int lda) {
  const int lane = threadIdx.x & 31;
  const __bf16* q = p + (size_t)(lane & 15) * lda + ((lane >> 4) << 4);
  V16U u;
  u.h[0] = *(const v8bf*)q;
  u.h[1] = *(const v8bf*)(q + 8);
  return u.v;
}

// B fragment (32x16) of B = W^T where W is row-major [N][K]; ptr at (n0,k0) of W.
// Layout used here (contiguous k, same as A): lane holds col N=lane&15; elements K = e + 16*(lane>=16) -> one
// contiguous 32B load per lane.
__device__ __forceinline__ v16bf load_b_frag(const __bf16* p, int ldw) {
  const int lane = threadIdx.x & 31;
  return *(const v16bf*)(p + (size_t)(lane & 15) * ldw + ((lane >> 4) << 4));
}

// CDNA5 async copy: 16B global -> LDS per lane, tracked by ASYNCcnt.
__device__ __forceinline__ void async_ld16(unsigned lds_off, const __bf16* g) {
  asm volatile("global_load_async_to_lds_b128 %0, %1, off"
               :: "v"(lds_off), "v"(g)
               : "memory");
}
__device__ __forceinline__ void wait_async0() {
  asm volatile("s_wait_asynccnt 0x0" ::: "memory");
}

// ---------------- fp32 -> bf16 convert ----------------
__global__ __launch_bounds__(256)
void cvt_kernel(const float* __restrict__ in, __bf16* __restrict__ out, int n) {
  int i = blockIdx.x * blockDim.x + threadIdx.x;
  const int stride = gridDim.x * blockDim.x;
  for (; i < n; i += stride) out[i] = (__bf16)in[i];
}

// ---------------- GEMM: C[M][N] = A[M][K] * W[N][K]^T ----------------
// SMODE 0: C bf16 row-major; 1: C bf16 transposed [N][M]; 2: C f32 row-major.
// Block: 256 threads = 8 waves (2 in M x 4 in N); block tile 128(M) x 256(N);
// wave tile 64x64 = 4x4 fragments. A tile (128x32) double-buffered in LDS via
// async-to-LDS copies; B (weights, L2-resident) streamed direct with prefetch.
template <int SMODE>
__global__ __launch_bounds__(256)
void gemm_bf16_kernel(const __bf16* __restrict__ A, const __bf16* __restrict__ W,
                      void* __restrict__ C, int M, int N, int K) {
  __shared__ __align__(16) __bf16 atile[2][128 * 32];

  const int tid = threadIdx.x;
  const int wave = tid >> 5;
  const int lane = tid & 31;
  const int wm = wave & 1;
  const int wn = wave >> 1;
  const int rowB = blockIdx.y * 128;       // block row base
  const int row0 = wm * 64;                // wave row base within block tile
  const int col0 = blockIdx.x * 256 + wn * 64;

  v8f acc[4][4];
#pragma unroll
  for (int i = 0; i < 4; ++i)
#pragma unroll
    for (int j = 0; j < 4; ++j) acc[i][j] = zero8();

  // cooperative async load of a 128x32 bf16 A tile (8KB): 512 x 16B chunks,
  // 2 per thread. LDS tile is row-major 128x32.
  auto issueA = [&](int buf, int k0) {
#pragma unroll
    for (int c = tid; c < 512; c += 256) {
      const int row = c >> 2;
      const int seg = c & 3;
      async_ld16((unsigned)(size_t)&atile[buf][row * 32 + seg * 8],
                 A + (size_t)(rowB + row) * K + k0 + seg * 8);
    }
  };

  const int niter = K >> 5;
  issueA(0, 0);
  for (int it = 0; it < niter; ++it) {
    wait_async0();        // this wave's copies into atile[it&1] are done
    __syncthreads();      // everyone's copies done; prev buffer reads done
    if (it + 1 < niter) issueA((it + 1) & 1, (it + 1) << 5);  // overlap w/ WMMA

    const __bf16* at = &atile[it & 1][row0 * 32];
    v16bf a[4], b[4];
#pragma unroll
    for (int i = 0; i < 4; ++i) a[i] = load_a_frag(at + i * 16 * 32, 32);
#pragma unroll
    for (int j = 0; j < 4; ++j) {
      const __bf16* wp = W + (size_t)(col0 + j * 16) * K + (it << 5);
      b[j] = load_b_frag(wp, K);
      __builtin_prefetch(wp + 128, 0, 1);  // ~4 K-iterations ahead
    }
#pragma unroll
    for (int i = 0; i < 4; ++i)
#pragma unroll
      for (int j = 0; j < 4; ++j)
        acc[i][j] = wmma_bf16(a[i], b[j], acc[i][j]);
  }

  // C/D layout: element r -> row = r + 8*(lane>=16), col = lane&15
  const int nIdx = lane & 15;
  const int rAdd = (lane >> 4) << 3;
#pragma unroll
  for (int i = 0; i < 4; ++i) {
#pragma unroll
    for (int j = 0; j < 4; ++j) {
      const int col = col0 + j * 16 + nIdx;
      if (SMODE == 1) {
        // transposed store: 8 consecutive rows -> one 16B vector store per lane
        v8bf pk;
#pragma unroll
        for (int r = 0; r < 8; ++r) pk[r] = (__bf16)acc[i][j][r];
        *(v8bf*)((__bf16*)C + (size_t)col * M + rowB + row0 + i * 16 + rAdd) = pk;
      } else {
#pragma unroll
        for (int r = 0; r < 8; ++r) {
          const int row = rowB + row0 + i * 16 + rAdd + r;
          if (SMODE == 0)
            ((__bf16*)C)[(size_t)row * N + col] = (__bf16)acc[i][j][r];
          else
            ((float*)C)[(size_t)row * N + col] = acc[i][j][r];
        }
      }
    }
  }
}

// ---------------- RoPE (half-split, theta=1e6), in-place on bf16 q/k ----------------
__global__ __launch_bounds__(256)
void rope_kernel(__bf16* __restrict__ qb, __bf16* __restrict__ kb,
                 const int* __restrict__ pos) {
  const int t = blockIdx.x;
  const float p = (float)pos[t];
  const float kexp = -19.9315685693f / 64.0f;  // -log2(1e6)/(HD/2)
  for (int u = threadIdx.x; u < (NH + NKV) * 64; u += blockDim.x) {
    __bf16* buf;
    size_t base;
    int i;
    if (u < NH * 64) {
      const int h = u >> 6; i = u & 63;
      buf = qb; base = ((size_t)t * NH + h) * HDIM;
    } else {
      const int uu = u - NH * 64;
      const int h = uu >> 6; i = uu & 63;
      buf = kb; base = ((size_t)t * NKV + h) * HDIM;
    }
    const float fr = p * exp2f((float)i * kexp);
    float sn, cs;
    __sincosf(fr, &sn, &cs);
    const float x1 = (float)buf[base + i];
    const float x2 = (float)buf[base + 64 + i];
    buf[base + i]      = (__bf16)(x1 * cs - x2 * sn);
    buf[base + 64 + i] = (__bf16)(x2 * cs + x1 * sn);
  }
}

// ---------------- causal GQA flash attention ----------------
// grid: (SEQ/64, NH); block 128 = 4 waves; each wave owns 16 query rows of one
// head, full HD=128 in registers. All 4 waves share one kv head -> K/V tiles
// are staged in LDS via CDNA5 async copies, double-buffered across key blocks.
__global__ __launch_bounds__(128)
void attn_kernel(const __bf16* __restrict__ qb, const __bf16* __restrict__ kb,
                 const __bf16* __restrict__ vtb, __bf16* __restrict__ ob) {
  const int head = blockIdx.y;
  const int g = head >> 2;  // NH/NKV = 4 query heads per kv head
  const int tid = threadIdx.x;
  const int wave = tid >> 5;
  const int lane = tid & 31;
  const int q0 = blockIdx.x * 64 + wave * 16;

  __shared__ __align__(16) __bf16 ktile[2][64 * 128];   // [key][hd]
  __shared__ __align__(16) __bf16 vtile[2][128 * 64];   // [hd][key]
  __shared__ __align__(16) __bf16 psh[4][16][64];

  // cooperative async stage of one 64-key K/V block (16KB each)
  auto issueKV = [&](int buf, int jj) {
#pragma unroll
    for (int c = tid; c < 1024; c += 128) {   // K: 64 rows x 16 segs
      const int row = c >> 4;
      const int seg = c & 15;
      async_ld16((unsigned)(size_t)&ktile[buf][row * 128 + seg * 8],
                 kb + (size_t)(jj + row) * NKVHD + g * HDIM + seg * 8);
    }
#pragma unroll
    for (int c = tid; c < 1024; c += 128) {   // V^T: 128 rows x 8 segs
      const int row = c >> 3;
      const int seg = c & 7;
      async_ld16((unsigned)(size_t)&vtile[buf][row * 64 + seg * 8],
                 vtb + (size_t)(g * HDIM + row) * SEQ + jj + seg * 8);
    }
  };

  // Q fragments: 4 chunks of 32 along head dim (kept in registers)
  v16bf qa[4];
#pragma unroll
  for (int kc = 0; kc < 4; ++kc)
    qa[kc] = load_a_frag(qb + (size_t)q0 * NHHD + head * HDIM + kc * 32, NHHD);

  v8f oacc[8];
#pragma unroll
  for (int nf2 = 0; nf2 < 8; ++nf2) oacc[nf2] = zero8();
  float mrow[8], lrow[8];
#pragma unroll
  for (int r = 0; r < 8; ++r) { mrow[r] = -1e30f; lrow[r] = 0.0f; }

  const float scale = 0.08838834764831845f;  // 1/sqrt(128)
  const int nIdx = lane & 15;
  const int rAdd = (lane >> 4) << 3;
  const int jend = blockIdx.x * 64 + 64;  // uniform across block -> barrier-safe

  issueKV(0, 0);
  for (int j = 0; j < jend; j += 64) {
    const int buf = (j >> 6) & 1;
    wait_async0();        // this wave's copies into buf are complete
    __syncthreads();      // all waves' copies done; buf^1 reads (iter-1) done
    if (j + 64 < jend) issueKV(buf ^ 1, j + 64);  // overlap with WMMA burst

    const __bf16* kt = &ktile[buf][0];
    const __bf16* vt = &vtile[buf][0];

    // S = Q @ K^T for 16 queries x 64 keys (K fragments from LDS)
    v8f s[4];
#pragma unroll
    for (int nf = 0; nf < 4; ++nf) s[nf] = zero8();
#pragma unroll
    for (int nf = 0; nf < 4; ++nf)
#pragma unroll
      for (int kc = 0; kc < 4; ++kc) {
        v16bf kf = load_b_frag(kt + (nf * 16) * HDIM + kc * 32, HDIM);
        s[nf] = wmma_bf16(qa[kc], kf, s[nf]);
      }

    // scale + causal mask
#pragma unroll
    for (int nf = 0; nf < 4; ++nf)
#pragma unroll
      for (int r = 0; r < 8; ++r) {
        const int qrow = q0 + rAdd + r;
        const int key = j + nf * 16 + nIdx;
        const float val = s[nf][r] * scale;
        s[nf][r] = (key <= qrow) ? val : -1e30f;
      }

    // online softmax per query row; P -> LDS (C-layout -> A-layout transit)
#pragma unroll
    for (int r = 0; r < 8; ++r) {
      float rm = fmaxf(fmaxf(s[0][r], s[1][r]), fmaxf(s[2][r], s[3][r]));
#pragma unroll
      for (int d = 1; d < 16; d <<= 1) rm = fmaxf(rm, __shfl_xor(rm, d, 32));
      const float mnew = fmaxf(mrow[r], rm);
      const float corr = __expf(mrow[r] - mnew);
      mrow[r] = mnew;
      float rsum = 0.0f;
#pragma unroll
      for (int nf = 0; nf < 4; ++nf) {
        const float pv = __expf(s[nf][r] - mnew);
        rsum += pv;
        psh[wave][rAdd + r][nf * 16 + nIdx] = (__bf16)pv;
      }
#pragma unroll
      for (int d = 1; d < 16; d <<= 1) rsum += __shfl_xor(rsum, d, 32);
      lrow[r] = lrow[r] * corr + rsum;
#pragma unroll
      for (int nf2 = 0; nf2 < 8; ++nf2) oacc[nf2][r] *= corr;
    }
    __syncthreads();

    // O += P @ V (P and V^T fragments both from LDS)
#pragma unroll
    for (int kc2 = 0; kc2 < 2; ++kc2) {
      v16bf pa = load_a_frag(&psh[wave][0][0] + kc2 * 32, 64);
#pragma unroll
      for (int nf2 = 0; nf2 < 8; ++nf2) {
        v16bf vf = load_b_frag(vt + (nf2 * 16) * 64 + kc2 * 32, 64);
        oacc[nf2] = wmma_bf16(pa, vf, oacc[nf2]);
      }
    }
    __syncthreads();
  }

  // normalize + store attention output (bf16, row-major [token][NH*HD])
#pragma unroll
  for (int nf2 = 0; nf2 < 8; ++nf2)
#pragma unroll
    for (int r = 0; r < 8; ++r) {
      const int row = q0 + rAdd + r;
      const int col = head * HDIM + nf2 * 16 + nIdx;
      ob[(size_t)row * NHHD + col] = (__bf16)(oacc[nf2][r] / lrow[r]);
    }
}

// ---------------- host launch ----------------
extern "C" void kernel_launch(void* const* d_in, const int* in_sizes, int n_in,
                              void* d_out, int out_size, void* d_ws, size_t ws_size,
                              hipStream_t stream) {
  (void)in_sizes; (void)n_in; (void)out_size; (void)ws_size;
  const float* h  = (const float*)d_in[0];
  const int*  pos = (const int*)d_in[1];
  const float* wq = (const float*)d_in[2];
  const float* wk = (const float*)d_in[3];
  const float* wv = (const float*)d_in[4];
  const float* wo = (const float*)d_in[5];
  float* out = (float*)d_out;

  char* ws = (char*)d_ws;
  size_t off = 0;
  auto alloc = [&](size_t elems) -> __bf16* {
    __bf16* p = (__bf16*)(ws + off);
    off += ((elems * sizeof(__bf16) + 255) / 256) * 256;
    return p;
  };
  __bf16* h16  = alloc((size_t)SEQ * HID);
  __bf16* wq16 = alloc((size_t)NHHD * HID);
  __bf16* wk16 = alloc((size_t)NKVHD * HID);
  __bf16* wv16 = alloc((size_t)NKVHD * HID);
  __bf16* wo16 = alloc((size_t)HID * NHHD);
  __bf16* q16  = alloc((size_t)SEQ * NHHD);
  __bf16* k16  = alloc((size_t)SEQ * NKVHD);
  __bf16* vt16 = alloc((size_t)NKVHD * SEQ);   // transposed V
  __bf16* a16  = alloc((size_t)SEQ * NHHD);
  // total workspace used: ~136 MB

  cvt_kernel<<<2048, 256, 0, stream>>>(h,  h16,  SEQ * HID);
  cvt_kernel<<<4096, 256, 0, stream>>>(wq, wq16, NHHD * HID);
  cvt_kernel<<<2048, 256, 0, stream>>>(wk, wk16, NKVHD * HID);
  cvt_kernel<<<2048, 256, 0, stream>>>(wv, wv16, NKVHD * HID);
  cvt_kernel<<<4096, 256, 0, stream>>>(wo, wo16, HID * NHHD);

  dim3 blk(256);
  gemm_bf16_kernel<0><<<dim3(NHHD / 256,  SEQ / 128), blk, 0, stream>>>(h16, wq16, q16,  SEQ, NHHD,  HID);
  gemm_bf16_kernel<0><<<dim3(NKVHD / 256, SEQ / 128), blk, 0, stream>>>(h16, wk16, k16,  SEQ, NKVHD, HID);
  gemm_bf16_kernel<1><<<dim3(NKVHD / 256, SEQ / 128), blk, 0, stream>>>(h16, wv16, vt16, SEQ, NKVHD, HID);

  rope_kernel<<<SEQ, 256, 0, stream>>>(q16, k16, pos);

  attn_kernel<<<dim3(SEQ / 64, NH), dim3(128), 0, stream>>>(q16, k16, vt16, a16);

  gemm_bf16_kernel<2><<<dim3(NHHD / 256, SEQ / 128), blk, 0, stream>>>(a16, wo16, out, SEQ, HID, NHHD);
}
